// RKDAngleLoss_12799002542289
// MI455X (gfx1250) — matrix operand order repaired and checked by reading, not compiled, not run on hardware
//
#include <hip/hip_runtime.h>
#include <math.h>

typedef __attribute__((ext_vector_type(16))) _Float16 v16h;
typedef __attribute__((ext_vector_type(8)))  _Float16 v8h;
typedef __attribute__((ext_vector_type(8)))  float    v8f;

#define NPTS 512
#define DIM  64
#define ROW_BYTES 144                       // 64 f16 + 8 f16 pad (conflict-free b128 tiles)
#define US_OFF 0
#define UT_OFF (NPTS * ROW_BYTES)           // 73728
#define RED_OFF (2 * NPTS * ROW_BYTES)      // 147456
#define CEN_OFF (RED_OFF + 256 * 4)         // 148480
#define SMEM_BYTES (CEN_OFF + 2 * DIM * 4)  // 148992 B: 2 workgroups fit in 320 KB/WGP

#define TILE_ROW_BYTES (16 * ROW_BYTES)     // 2304
#define BSTEP (8 * 16 * ROW_BYTES)          // column-tile stride per wave step (18432)

// A-operand: per lane-half, contiguous 32 B (two b128 at +0/+16), k = 16h + s
__device__ __forceinline__ v16h lds_load_a(const char* base, int off) {
  v8h lo = *(const v8h*)(base + off);
  v8h hi = *(const v8h*)(base + off + 16);
  return __builtin_shufflevector(lo, hi, 0, 1, 2, 3, 4, 5, 6, 7, 8, 9, 10, 11, 12, 13, 14, 15);
}

// B-operand (B = U^T): per lane-half, contiguous 32 B (two b128 at +0/+16)
__device__ __forceinline__ v16h lds_load_b(const char* base, int off) {
  v8h lo = *(const v8h*)(base + off);
  v8h hi = *(const v8h*)(base + off + 16);
  return __builtin_shufflevector(lo, hi, 0, 1, 2, 3, 4, 5, 6, 7, 8, 9, 10, 11, 12, 13, 14, 15);
}

// branchless smooth-L1: m = min(|d|,1); loss = 0.5*m*(2|d| - m); hw = 0.5*weight
__device__ __forceinline__ void smooth_l1_acc(const v8f& cs, const v8f& ct, float hw, float& lacc) {
#pragma unroll
  for (int k = 0; k < 8; ++k) {
    float d  = cs[k] - ct[k];
    float ad = fabsf(d);
    float m  = fminf(ad, 1.0f);
    float t  = fmaf(2.0f, ad, -m);
    lacc = fmaf(hw, m * t, lacc);
  }
}

// normalized diff row (f32 math, eps clamp as in reference) -> f16 LDS row
__device__ __forceinline__ void norm_row_to_lds(const float* __restrict__ row,
                                                const float* cen, char* dst) {
  float d[DIM];
  float ss = 0.f;
#pragma unroll
  for (int q = 0; q < DIM / 4; ++q) {
    float4 e = ((const float4*)row)[q];
    float4 c = ((const float4*)cen)[q];
    float x = e.x - c.x, y = e.y - c.y, z = e.z - c.z, w = e.w - c.w;
    d[4 * q + 0] = x; d[4 * q + 1] = y; d[4 * q + 2] = z; d[4 * q + 3] = w;
    ss += x * x + y * y + z * z + w * w;
  }
  float scale = 1.0f / fmaxf(sqrtf(ss), 1e-12f);
#pragma unroll
  for (int q = 0; q < 8; ++q) {
    v8h h;
#pragma unroll
    for (int k = 0; k < 8; ++k) h[k] = (_Float16)(d[8 * q + k] * scale);
    *(v8h*)(dst + 16 * q) = h;   // ds_store_b128
  }
}

// 4 WMMAs for one 16x16 tile pair (student + teacher), K = 64 in two steps
__device__ __forceinline__ void tile_gram(const char* us, const char* ut, int boff,
                                          const v16h& as0, const v16h& as1,
                                          const v16h& at0, const v16h& at1,
                                          v8f& cs, v8f& ct) {
  v16h bs0 = lds_load_b(us, boff);
  v16h bs1 = lds_load_b(us, boff + 64);
  v16h bt0 = lds_load_b(ut, boff);
  v16h bt1 = lds_load_b(ut, boff + 64);
  cs = __builtin_amdgcn_wmma_f32_16x16x32_f16(false, as0, false, bs0, (short)0, cs, false, false);
  cs = __builtin_amdgcn_wmma_f32_16x16x32_f16(false, as1, false, bs1, (short)0, cs, false, false);
  ct = __builtin_amdgcn_wmma_f32_16x16x32_f16(false, at0, false, bt0, (short)0, ct, false, false);
  ct = __builtin_amdgcn_wmma_f32_16x16x32_f16(false, at1, false, bt1, (short)0, ct, false, false);
}

__global__ void rkd_angle_main(const float* __restrict__ S, const float* __restrict__ T,
                               float* __restrict__ part) {
  extern __shared__ char smem[];
  const int i     = blockIdx.x;
  const int tid   = threadIdx.x;
  const int lane  = tid & 31;
  const int wave  = tid >> 5;
  const int lrow  = lane & 15;
  const int lhalf = lane >> 4;

  char* us = smem + US_OFF;
  char* ut = smem + UT_OFF;
  float* red = (float*)(smem + RED_OFF);
  float* cen = (float*)(smem + CEN_OFF);

  // ---- Phase 0: anchor rows into LDS ----
  if (tid < DIM) {
    cen[tid]       = S[i * DIM + tid];
    cen[DIM + tid] = T[i * DIM + tid];
  }
  __syncthreads();

  // ---- Phase 1: build normalized diff matrices (f16) in LDS ----
#pragma unroll
  for (int rr = 0; rr < 2; ++rr) {
    int j = tid + rr * 256;
    norm_row_to_lds(S + j * DIM, cen,       us + j * ROW_BYTES);
    norm_row_to_lds(T + j * DIM, cen + DIM, ut + j * ROW_BYTES);
  }
  __syncthreads();

  // ---- Phase 2: Gram tiles via WMMA; symmetry => only tc >= tr, off-diag x2 ----
  float lacc = 0.f;
  int aoff = lrow * ROW_BYTES + (lhalf << 5);   // A byte offset, stepped per tile-row
  for (int tr = 0; tr < 32; ++tr, aoff += TILE_ROW_BYTES) {
    v16h as0 = lds_load_a(us, aoff);
    v16h as1 = lds_load_a(us, aoff + 64);
    v16h at0 = lds_load_a(ut, aoff);
    v16h at1 = lds_load_a(ut, aoff + 64);

    int tc   = tr + wave;
    int boff = (tc * 16 + lrow) * ROW_BYTES + (lhalf << 5);

    // pairs of column tiles: independent accumulators fill WMMA hazard slots
    while (tc + 8 < 32) {
      float hw0 = (tc == tr) ? 0.5f : 1.0f;   // 0.5 * weight
      v8f cs0 = {}, ct0 = {}, cs1 = {}, ct1 = {};
      tile_gram(us, ut, boff,         as0, as1, at0, at1, cs0, ct0);
      tile_gram(us, ut, boff + BSTEP, as0, as1, at0, at1, cs1, ct1);
      smooth_l1_acc(cs0, ct0, hw0,  lacc);
      smooth_l1_acc(cs1, ct1, 1.0f, lacc);   // tc+8 > tr always
      tc += 16;
      boff += 2 * BSTEP;
    }
    if (tc < 32) {
      float hw0 = (tc == tr) ? 0.5f : 1.0f;
      v8f cs0 = {}, ct0 = {};
      tile_gram(us, ut, boff, as0, as1, at0, at1, cs0, ct0);
      smooth_l1_acc(cs0, ct0, hw0, lacc);
    }
  }

  // ---- Phase 3: block reduction, deterministic partial out ----
  __syncthreads();
  red[tid] = lacc;
  __syncthreads();
  for (int s2 = 128; s2 > 0; s2 >>= 1) {
    if (tid < s2) red[tid] += red[tid + s2];
    __syncthreads();
  }
  if (tid == 0) part[i] = red[0];
}

__global__ void rkd_angle_reduce(const float* __restrict__ part, float* __restrict__ out) {
  __shared__ float red[256];
  int tid = threadIdx.x;
  red[tid] = part[tid] + part[tid + 256];
  __syncthreads();
  for (int s2 = 128; s2 > 0; s2 >>= 1) {
    if (tid < s2) red[tid] += red[tid + s2];
    __syncthreads();
  }
  if (tid == 0) out[0] = red[0] * (1.0f / 134217728.0f);  // mean over 512^3
}

extern "C" void kernel_launch(void* const* d_in, const int* in_sizes, int n_in,
                              void* d_out, int out_size, void* d_ws, size_t ws_size,
                              hipStream_t stream) {
  const float* S = (const float*)d_in[0];  // student_emb [512,64] f32
  const float* T = (const float*)d_in[1];  // teacher_emb [512,64] f32
  float* out  = (float*)d_out;             // scalar f32
  float* part = (float*)d_ws;              // 512 block partials (2 KB)

  rkd_angle_main<<<dim3(NPTS), dim3(256), SMEM_BYTES, stream>>>(S, T, part);
  rkd_angle_reduce<<<dim3(1), dim3(256), 0, stream>>>(part, out);
}
